// MultiHeadAttention_71880572666209
// MI455X (gfx1250) — matrix operand order repaired and checked by reading, not compiled, not run on hardware
//
#include <hip/hip_runtime.h>

// ---------------------------------------------------------------------------
// MI455X (gfx1250) multi-head attention, bf16 WMMA + flash-softmax fusion.
// Scores are computed TRANSPOSED (S^T = K*Q^T) so the score C-tile layout
// coincides with the A-fragment layout needed by the P*V WMMA: no LDS
// transpose for P, per-lane softmax statistics (one query per lane), and with
// V pre-transposed per head the flash loop is barrier-free and LDS-free.
// ---------------------------------------------------------------------------

typedef __attribute__((ext_vector_type(16))) __bf16 v16bf;
typedef __attribute__((ext_vector_type(8)))  __bf16 v8bf;
typedef __attribute__((ext_vector_type(8)))  float  v8f;
typedef __attribute__((ext_vector_type(4)))  float  f32x4;

#define HIDDEN 2048
#define NHEADS 32
#define HDIM   16
#define PROJ   512
#define SEQ    2048
#define BATCH  2
#define SCALE  0.125f   // (2048/8/4)^-0.5

// ---- helpers --------------------------------------------------------------

__device__ inline v8f zero8f() {
  v8f z;
#pragma unroll
  for (int i = 0; i < 8; ++i) z[i] = 0.0f;
  return z;
}

__device__ inline v16bf zero16bf() {
  v16bf z;
#pragma unroll
  for (int i = 0; i < 16; ++i) z[i] = (__bf16)0.0f;
  return z;
}

__device__ inline v16bf frag_cat(v8bf lo, v8bf hi) {
  v16bf r;
#pragma unroll
  for (int i = 0; i < 8; ++i) { r[i] = lo[i]; r[i + 8] = hi[i]; }
  return r;
}

__device__ inline v8f wmma_bf16(v16bf a, v16bf b, v8f c) {
  // D = A(16x32 bf16) * B(32x16 bf16) + C(16x16 f32)
  return __builtin_amdgcn_wmma_f32_16x16x32_bf16(
      /*neg_a=*/false, a, /*neg_b=*/false, b,
      /*c_mod=*/(short)0, c, /*reuse_a=*/false, /*reuse_b=*/false);
}

// ---- kernel 1: f32 -> bf16 weight conversion (scale folds q-scaling) ------

__global__ __launch_bounds__(256) void cvt_bf16_kernel(
    const float* __restrict__ src, __bf16* __restrict__ dst, int n, float scale) {
  int i = blockIdx.x * 256 + threadIdx.x;
  if (i < n) dst[i] = (__bf16)(src[i] * scale);
}

// ---- kernels 2 & 5: WMMA GEMM  C[M,N] = A[M,K] * W[N,K]^T ------------------
// A: f32 (converted in-flight) or bf16.  W: bf16 row-major [N][K].
// Block = 256 thr = 8 waves; wave computes a 16x64 output tile (4 WMMA tiles
// sharing one A fragment per k-step).  grid = (M/16, N/512)

template <bool AF32, bool OUTBF>
__global__ __launch_bounds__(256) void gemm16_kernel(
    const void* __restrict__ Ap, const __bf16* __restrict__ W,
    void* __restrict__ Cp, int M, int N, int K) {
  const int wave = threadIdx.x >> 5;
  const int lane = threadIdx.x & 31;
  const int half = lane >> 4;   // which 16-lane half of the wave
  const int lc   = lane & 15;

  const int m0  = blockIdx.x * 16;
  const int n0  = blockIdx.y * 512 + wave * 64;
  const int row = m0 + lc;

  v8f acc[4];
#pragma unroll
  for (int t = 0; t < 4; ++t) acc[t] = zero8f();

  for (int k = 0; k < K; k += 32) {
    // A fragment: lane holds row `lc`; K chunks {base..base+7, base+16..base+23}
    const int abase = k + half * 8;
    v16bf afrag;
    if constexpr (AF32) {
      const float* ap = (const float*)Ap + (size_t)row * K + abase;
      f32x4 x0 = *(const f32x4*)(ap + 0);
      f32x4 x1 = *(const f32x4*)(ap + 4);
      f32x4 x2 = *(const f32x4*)(ap + 16);
      f32x4 x3 = *(const f32x4*)(ap + 20);
#pragma unroll
      for (int i = 0; i < 4; ++i) {
        afrag[i]      = (__bf16)x0[i];
        afrag[4 + i]  = (__bf16)x1[i];
        afrag[8 + i]  = (__bf16)x2[i];
        afrag[12 + i] = (__bf16)x3[i];
      }
    } else {
      const __bf16* ap = (const __bf16*)Ap + (size_t)row * K + abase;
      afrag = frag_cat(*(const v8bf*)ap, *(const v8bf*)(ap + 16));
    }

    // B fragments: lane holds column n, K chunks {kb..kb+7, kb+16..kb+23}
    const int kb = k + half * 8;
#pragma unroll
    for (int t = 0; t < 4; ++t) {
      const __bf16* bp = W + (size_t)(n0 + t * 16 + lc) * K + kb;
      v16bf bfrag = frag_cat(*(const v8bf*)bp, *(const v8bf*)(bp + 16));
      acc[t] = wmma_bf16(afrag, bfrag, acc[t]);
    }
  }

  // C tile: lane<16 -> rows 0..7 (vgpr r = row r), lane>=16 -> rows 8..15
#pragma unroll
  for (int r = 0; r < 8; ++r) {
    const int    orow = m0 + r + half * 8;
    const size_t o0   = (size_t)orow * N + n0 + lc;
#pragma unroll
    for (int t = 0; t < 4; ++t) {
      if constexpr (OUTBF) {
        ((__bf16*)Cp)[o0 + t * 16] = (__bf16)acc[t][r];
      } else {
        ((float*)Cp)[o0 + t * 16] = acc[t][r];
      }
    }
  }
}

// ---- kernel 3: per-head V transpose via async global->LDS copy ------------
// Vh  [B*SEQ][PROJ]  (bf16, row-major per token)
//   -> VhT [(b*NH+h)*HDIM + d][SEQ]  (bf16, contiguous along sequence)
// Rows are DMA'd into LDS with GLOBAL_LOAD_ASYNC_TO_LDS_B128 (ASYNCcnt),
// then written back transposed with coalesced 16B stores.
// grid = (B*NH, SEQ/128), block = 256.

__global__ __launch_bounds__(256) void transpose_v_kernel(
    const __bf16* __restrict__ Vh, __bf16* __restrict__ VhT) {
  __shared__ __bf16 T[128][16];   // [key][d], rows are 32B

  const int bh = blockIdx.x;          // b*NHEADS + h
  const int b  = bh >> 5;
  const int h  = bh & 31;
  const int k0 = blockIdx.y * 128;
  const int t  = threadIdx.x;

  // Async-copy 128 key-rows x 32B (16 bf16) into LDS: 2 x b128 per row.
  {
    const int key  = t >> 1;
    const int part = (t & 1) * 8;   // which 16B half of the 32B row
    const __bf16* src =
        Vh + ((size_t)(b * SEQ + k0 + key)) * PROJ + (size_t)h * HDIM + part;
    const unsigned ldsoff = (unsigned)(uintptr_t)(&T[key][part]);
    asm volatile("global_load_async_to_lds_b128 %0, %1, off"
                 :: "v"(ldsoff), "v"(src) : "memory");
  }
  asm volatile("s_wait_asynccnt 0x0" ::: "memory");
  __syncthreads();

  // Coalesced transposed write-back: thread -> (d, 8 consecutive keys).
  {
    const int d  = t >> 4;          // 0..15
    const int kb = (t & 15) * 8;    // 0,8,..,120
    v8bf o;
#pragma unroll
    for (int j = 0; j < 8; ++j) o[j] = T[kb + j][d];
    *(v8bf*)(VhT + ((size_t)bh * HDIM + d) * SEQ + k0 + kb) = o;
  }
}

// ---- kernel 4: fused flash attention (barrier-free, LDS-free) -------------
// grid = (B*NH, SEQ/128); block = 256 thr = 8 waves; wave owns 16 query rows.
// Per 32-key step: two S^T = K*Q^T WMMAs (head dim 16 zero-padded to K=32),
// per-lane online softmax (lane <-> query), P built in registers, one
// full-K=32 P*V WMMA with V fragments loaded straight from pre-transposed VhT.

__global__ __launch_bounds__(256) void attn_flash_kernel(
    const __bf16* __restrict__ Qh, const __bf16* __restrict__ Kh,
    const __bf16* __restrict__ VhT, __bf16* __restrict__ Ob) {
  const int b    = blockIdx.x >> 5;   // /NHEADS
  const int h    = blockIdx.x & 31;
  const int q0   = blockIdx.y * 128;
  const int wave = threadIdx.x >> 5;
  const int lane = threadIdx.x & 31;
  const int half = lane >> 4;
  const int lc   = lane & 15;

  const size_t headoff = (size_t)h * HDIM;

  // Q as B operand (32d x 16 queries): elems 0..7 carry d=half*8..half*8+7;
  // elems 8..15 are the d=16..31 padding.
  v16bf qfrag = zero16bf();
  {
    const __bf16* qp =
        Qh + ((size_t)(b * SEQ + q0 + wave * 16 + lc)) * PROJ + headoff + half * 8;
    v8bf qv = *(const v8bf*)qp;
#pragma unroll
    for (int i = 0; i < 8; ++i) qfrag[i] = qv[i];
  }

  // Per-lane V^T stream: lane = d column `lc`; keys are contiguous.
  const __bf16* vrow =
      VhT + ((size_t)blockIdx.x * HDIM + lc) * SEQ + half * 8;

  float m = -__builtin_inff();  // running max for query `lc` (lane pair)
  float l = 0.0f;               // running sum for query `lc`
  v8f acc = zero8f();           // O tile: lane=d column, vgpr r = query r+8*half

  for (int jj = 0; jj < SEQ; jj += 32) {
    // K as A operand (two 16-key tiles).  Every lane loads 8 real d-values
    // (elems 0..7), elems 8..15 are the d=16..31 zero padding.
    const __bf16* kp0 =
        Kh + ((size_t)(b * SEQ + jj + lc)) * PROJ + headoff + half * 8;
    const __bf16* kp1 = kp0 + (size_t)16 * PROJ;
    if (jj + 32 < SEQ) {
      __builtin_prefetch(kp0 + (size_t)32 * PROJ, 0, 0);  // next K tile
      __builtin_prefetch(vrow + jj + 32, 0, 0);           // next V^T chunk
    }
    v16bf ka0 = zero16bf(), ka1 = zero16bf();
    {
      v8bf k0 = *(const v8bf*)kp0;
      v8bf k1 = *(const v8bf*)kp1;
#pragma unroll
      for (int i = 0; i < 8; ++i) { ka0[i] = k0[i]; ka1[i] = k1[i]; }
    }

    // V as B fragment: lane = d column lc, keys jj+half*8 +{0..7, 16..23}.
    v16bf vfrag =
        frag_cat(*(const v8bf*)(vrow + jj), *(const v8bf*)(vrow + jj + 16));

    // S^T tiles: rows = keys, cols = queries.  Lane holds query `lc`;
    // vgpr i = key i (+8 for upper half-wave) of each tile.
    v8f s0 = wmma_bf16(ka0, qfrag, zero8f());
    v8f s1 = wmma_bf16(ka1, qfrag, zero8f());

    // ---- per-lane online softmax for query `lc` ----
    float t = fmaxf(s0[0], s1[0]);
#pragma unroll
    for (int i = 1; i < 8; ++i) t = fmaxf(t, fmaxf(s0[i], s1[i]));
    t = fmaxf(t, __shfl_xor(t, 16));          // merge lane pair (lc, lc+16)
    const float mn    = fmaxf(m, t);
    const float alpha = __expf(m - mn);
    m = mn;

    // P fragment built directly in registers: the S^T C-layout key indices
    // per lane match the A-fragment key chunks exactly.
    v16bf pfrag;
    float ps = 0.0f;
#pragma unroll
    for (int i = 0; i < 8; ++i) {
      const float p0 = __expf(s0[i] - mn);
      const float p1 = __expf(s1[i] - mn);
      ps += p0 + p1;
      pfrag[i]     = (__bf16)p0;
      pfrag[8 + i] = (__bf16)p1;
    }
    ps += __shfl_xor(ps, 16);                 // merge lane pair
    l = l * alpha + ps;

    // Rescale O: row r+8*half needs alpha of that query, which lives at
    // group-lane index r+8*half in each 16-lane half.
#pragma unroll
    for (int r = 0; r < 8; ++r) acc[r] *= __shfl(alpha, r + 8 * half, 16);

    acc = wmma_bf16(pfrag, vfrag, acc);
  }

  // Normalize (gather each row's l) and store O tile (bf16) to workspace.
#pragma unroll
  for (int r = 0; r < 8; ++r) {
    const float lf  = __shfl(l, r + 8 * half, 16);
    const int   row = q0 + wave * 16 + r + half * 8;
    Ob[((size_t)(b * SEQ + row)) * PROJ + headoff + lc] =
        (__bf16)(acc[r] / lf);
  }
}

// ---- launcher -------------------------------------------------------------

extern "C" void kernel_launch(void* const* d_in, const int* in_sizes, int n_in,
                              void* d_out, int out_size, void* d_ws, size_t ws_size,
                              hipStream_t stream) {
  const float* q  = (const float*)d_in[0];
  const float* k  = (const float*)d_in[1];
  const float* v  = (const float*)d_in[2];
  const float* Wq = (const float*)d_in[3];
  const float* Wk = (const float*)d_in[4];
  const float* Wv = (const float*)d_in[5];
  const float* Wo = (const float*)d_in[6];
  float* out = (float*)d_out;

  const int WN = PROJ * HIDDEN;          // 1,048,576 elements per weight
  const int XN = BATCH * SEQ * PROJ;     // 2,097,152 elements per activation

  __bf16* p    = (__bf16*)d_ws;
  __bf16* Wq_b = p; p += WN;
  __bf16* Wk_b = p; p += WN;
  __bf16* Wv_b = p; p += WN;
  __bf16* Wo_b = p; p += WN;
  __bf16* Qh   = p; p += XN;
  __bf16* Khp  = p; p += XN;
  __bf16* Vhp  = p; p += XN;
  __bf16* VhT  = p; p += XN;
  __bf16* Obuf = p;

  // 1) weights -> bf16 (fold score scale into Wq)
  const int cb = WN / 256;
  cvt_bf16_kernel<<<cb, 256, 0, stream>>>(Wq, Wq_b, WN, SCALE);
  cvt_bf16_kernel<<<cb, 256, 0, stream>>>(Wk, Wk_b, WN, 1.0f);
  cvt_bf16_kernel<<<cb, 256, 0, stream>>>(Wv, Wv_b, WN, 1.0f);
  cvt_bf16_kernel<<<cb, 256, 0, stream>>>(Wo, Wo_b, WN, 1.0f);

  // 2) Q/K/V projections: [4096,2048] x [2048,512] -> bf16
  dim3 gp(BATCH * SEQ / 16, PROJ / 512);
  gemm16_kernel<true, true><<<gp, 256, 0, stream>>>(q, Wq_b, Qh,
                                                    BATCH * SEQ, PROJ, HIDDEN);
  gemm16_kernel<true, true><<<gp, 256, 0, stream>>>(k, Wk_b, Khp,
                                                    BATCH * SEQ, PROJ, HIDDEN);
  gemm16_kernel<true, true><<<gp, 256, 0, stream>>>(v, Wv_b, Vhp,
                                                    BATCH * SEQ, PROJ, HIDDEN);

  // 3) per-head V transpose (async global->LDS DMA + coalesced write-back)
  dim3 gt(BATCH * NHEADS, SEQ / 128);
  transpose_v_kernel<<<gt, 256, 0, stream>>>(Vhp, VhT);

  // 4) fused flash attention per (batch, head)
  dim3 ga(BATCH * NHEADS, SEQ / 128);
  attn_flash_kernel<<<ga, 256, 0, stream>>>(Qh, Khp, VhT, Obuf);

  // 5) output projection: [4096,512] x [512,2048] -> f32 d_out
  dim3 go(BATCH * SEQ / 16, HIDDEN / 512);
  gemm16_kernel<false, false><<<go, 256, 0, stream>>>(Obuf, Wo_b, out,
                                                      BATCH * SEQ, HIDDEN, PROJ);
}
